// Trans4map_deformable_detr_68805376082046
// MI455X (gfx1250) — matrix operand order repaired and checked by reading, not compiled, not run on hardware
//
#include <hip/hip_runtime.h>
#include <hip/hip_bf16.h>

// =====================================================================
// Trans4map deformable-DETR pipeline for MI455X (gfx1250, wave32, WMMA)
//
// All GEMMs (encoder projections, FFNs, conv decoder as implicit GEMM)
// run on v_wmma_f32_16x16x32_f16 with f32 accumulate. GEMM A operands
// are produced in f16 by upstream kernels; weights are pre-converted to
// f16 [N][Kp] so B fragments load straight from global as b128 in the
// ISA 7.12.2 lane layout. Dense A tiles stage into LDS via the gfx1250
// async global->LDS copies (ASYNCcnt).
//
// Workspace requirement: ~300 MB.
// =====================================================================

typedef __attribute__((ext_vector_type(16))) _Float16 v16h;
typedef __attribute__((ext_vector_type(8)))  float    v8f;
typedef __attribute__((ext_vector_type(8)))  _Float16 h8;
typedef int v4i_t __attribute__((vector_size(16)));

#define AS1 __attribute__((address_space(1)))
#define AS3 __attribute__((address_space(3)))

#define NQv    22500
#define EMBEDv 256
#define NVALv  43520
#define MAPPIX 250000   // 500*500

#if defined(__HIP_DEVICE_COMPILE__) && \
    __has_builtin(__builtin_amdgcn_global_load_async_to_lds_b128) && \
    __has_builtin(__builtin_amdgcn_s_wait_asynccnt)
#define USE_ASYNC_LDS 1
#else
#define USE_ASYNC_LDS 0
#endif

// ------------------------------------------------------------------
// A-tile providers (all produce f16, branchless clamp+select guards)
// ------------------------------------------------------------------
struct DenseAH {               // A[M][K] row-major f16, K multiple of 32
  static constexpr bool RAW = true;     // plain strided copy -> async DMA ok
  const _Float16* A; int M; int K;
  __device__ const _Float16* rawPtr(int m, int k) const {
    int mc = m < M ? m : M - 1;         // clamped: rows >= M feed unsaved C rows
    return A + (size_t)mc * K + k;
  }
  __device__ h8 load8(int m, int k) const {
    int mc = m < M ? m : M - 1;
    h8 v = *reinterpret_cast<const h8*>(A + (size_t)mc * K + k);
    h8 z = {};
    return (m < M) ? v : z;
  }
  __device__ void prefetch(int m, int k) const {
    int mc = m < M ? m : M - 1;
    int kc = k < K ? k : K - 1;
    __builtin_prefetch(A + (size_t)mc * K + kc, 0, 1);  // global_prefetch_b8
  }
};

struct ConvA7H {               // im2col 7x7 over f16 CHW [256][500][500]
  static constexpr bool RAW = false;
  const _Float16* x;
  __device__ const _Float16* rawPtr(int, int) const { return x; }
  __device__ _Float16 get(int m, int k) const {
    int c = k / 49; int r = k - c * 49;
    int ky = r / 7; int kx = r - ky * 7;
    int mc = m < MAPPIX ? m : MAPPIX - 1;
    int oy = mc / 500, ox = mc - oy * 500;
    int iy = oy + ky - 3, ix = ox + kx - 3;
    bool ok = (m < MAPPIX) & ((unsigned)iy < 500u) & ((unsigned)ix < 500u);
    int iyc = iy < 0 ? 0 : (iy > 499 ? 499 : iy);
    int ixc = ix < 0 ? 0 : (ix > 499 ? 499 : ix);
    _Float16 v = x[(size_t)c * MAPPIX + iyc * 500 + ixc];
    return ok ? v : (_Float16)0.f;
  }
  __device__ h8 load8(int m, int k) const {
    h8 v;
    #pragma unroll
    for (int j = 0; j < 8; ++j) v[j] = get(m, k + j);
    return v;
  }
  __device__ void prefetch(int, int) const {}
};

struct ConvA3H {               // im2col 3x3 with fused input BN+ReLU
  static constexpr bool RAW = false;
  const _Float16* x; const float* mean; const float* rstd;
  const float* g; const float* b; int Cin;
  __device__ const _Float16* rawPtr(int, int) const { return x; }
  __device__ _Float16 get(int m, int k) const {
    int c = k / 9; int r = k - c * 9;
    int ky = r / 3; int kx = r - ky * 3;
    int mc = m < MAPPIX ? m : MAPPIX - 1;
    int oy = mc / 500, ox = mc - oy * 500;
    int iy = oy + ky - 1, ix = ox + kx - 1;
    bool ok = (m < MAPPIX) & (c < Cin) & ((unsigned)iy < 500u) & ((unsigned)ix < 500u);
    int cc = c < Cin ? c : Cin - 1;
    int iyc = iy < 0 ? 0 : (iy > 499 ? 499 : iy);
    int ixc = ix < 0 ? 0 : (ix > 499 ? 499 : ix);
    float v = (float)x[(size_t)cc * MAPPIX + iyc * 500 + ixc];
    v = (v - mean[cc]) * rstd[cc] * g[cc] + b[cc];
    v = v > 0.f ? v : 0.f;
    return ok ? (_Float16)v : (_Float16)0.f;
  }
  __device__ h8 load8(int m, int k) const {
    h8 v;
    #pragma unroll
    for (int j = 0; j < 8; ++j) v[j] = get(m, k + j);
    return v;
  }
  __device__ void prefetch(int, int) const {}
};

// ------------------------------------------------------------------
// WMMA GEMM core: tile 128(M) x 64(N), BK=32, 8 waves (4x2 of 32x32).
// B is f16 [N][K] in global memory, fragments loaded directly as b128.
// OUTMODE 0: f32 [M][N] out, + optional bias / residual / relu
// OUTMODE 1: f16 channel-major [N][MAPPIX] out (conv feature maps)
// OUTMODE 2: f16 row-major [M][N] out, + bias / relu (FFN hidden)
// ------------------------------------------------------------------
template <class AP, int OUTMODE>
__global__ __launch_bounds__(256)
void wmma_gemm(AP ap, const _Float16* __restrict__ Bw, int M, int N, int K,
               const float* __restrict__ bias,
               const float* __restrict__ residual,
               float* __restrict__ outF,
               _Float16* __restrict__ outH,
               int relu)
{
  __shared__ __align__(16) _Float16 As[128 * 40];  // [row][k] pad 40

  const int tid  = threadIdx.x;
  const int lane = tid & 31;
  const int wave = tid >> 5;
  const int wm   = (wave & 3) * 32;
  const int wn   = (wave >> 2) * 32;
  const int bm   = blockIdx.x * 128;
  const int bn   = blockIdx.y * 64;
  const int half = lane >> 4;
  const int l15  = lane & 15;

  const int arow = tid >> 1;          // A staging: 2 threads per row
  const int akb  = (tid & 1) * 16;

  v8f acc[2][2] = {};

  const int kTiles = K >> 5;          // K is padded to a multiple of 32
  for (int kt = 0; kt < kTiles; ++kt) {
    const int k0 = kt << 5;
    // ---- stage A tile ----
#if USE_ASYNC_LDS
    if constexpr (AP::RAW) {
      // async global->LDS DMA: 2x b128 per thread, tracked by ASYNCcnt
      const _Float16* src = ap.rawPtr(bm + arow, k0 + akb);
      __builtin_amdgcn_global_load_async_to_lds_b128(
          (AS1 v4i_t*)src, (AS3 v4i_t*)&As[arow * 40 + akb], 0, 0);
      __builtin_amdgcn_global_load_async_to_lds_b128(
          (AS1 v4i_t*)src, (AS3 v4i_t*)&As[arow * 40 + akb], 16, 0);
      ap.prefetch(bm + arow, k0 + 64 + akb);
      __builtin_amdgcn_s_wait_asynccnt(0);
    } else
#endif
    {
      h8 v0 = ap.load8(bm + arow, k0 + akb);
      h8 v1 = ap.load8(bm + arow, k0 + akb + 8);
      *reinterpret_cast<h8*>(&As[arow * 40 + akb])     = v0;
      *reinterpret_cast<h8*>(&As[arow * 40 + akb + 8]) = v1;
      ap.prefetch(bm + arow, k0 + 64 + akb);
    }
    __syncthreads();

    // ---- A fragments from LDS (ISA 7.12.2 16-bit A layout) ----
    v16h af[2], bf[2];
    #pragma unroll
    for (int mi = 0; mi < 2; ++mi) {
      const _Float16* ap0 = &As[(wm + mi * 16 + l15) * 40 + 8 * half];
      #pragma unroll
      for (int i = 0; i < 8; ++i) af[mi][i] = ap0[i];          // K = 8*half + i
      #pragma unroll
      for (int i = 0; i < 8; ++i) af[mi][8 + i] = ap0[16 + i]; // K = 16 + 8*half + i
    }
    // ---- B fragments straight from global (f16 [N][K], contiguous K) ----
    #pragma unroll
    for (int ni = 0; ni < 2; ++ni) {
      int coln = bn + wn + ni * 16 + l15;
      if (coln >= N) coln = N - 1;                    // clamp; stores guarded
      const _Float16* bp0 = Bw + (size_t)coln * K + k0 + 8 * half;
      h8 b0 = *reinterpret_cast<const h8*>(bp0);
      h8 b1 = *reinterpret_cast<const h8*>(bp0 + 16);
      #pragma unroll
      for (int i = 0; i < 8; ++i) { bf[ni][i] = b0[i]; bf[ni][8 + i] = b1[i]; }
    }
    #pragma unroll
    for (int mi = 0; mi < 2; ++mi)
      #pragma unroll
      for (int ni = 0; ni < 2; ++ni)
        acc[mi][ni] = __builtin_amdgcn_wmma_f32_16x16x32_f16(
            false, af[mi], false, bf[ni], (short)0, acc[mi][ni], false, false);
    __syncthreads();
  }

  // ---- store (full-tile fast path; partial tiles guarded) ----
  const bool fullM = (bm + 128 <= M);
  #pragma unroll
  for (int mi = 0; mi < 2; ++mi) {
    #pragma unroll
    for (int ni = 0; ni < 2; ++ni) {
      const int col   = bn + wn + ni * 16 + l15;
      const int rbase = bm + wm + mi * 16 + half * 8;
      if (col < N) {
        if (OUTMODE == 0) {
          const float bv = bias ? bias[col] : 0.f;
          float* op = outF + (size_t)rbase * N + col;
          const float* rp = residual ? residual + (size_t)rbase * N + col : nullptr;
          if (fullM) {
            #pragma unroll
            for (int r = 0; r < 8; ++r) {
              float v = acc[mi][ni][r] + bv;
              if (rp) v += rp[(size_t)r * N];
              if (relu) v = fmaxf(v, 0.f);
              op[(size_t)r * N] = v;
            }
          } else {
            #pragma unroll
            for (int r = 0; r < 8; ++r) {
              if (rbase + r < M) {
                float v = acc[mi][ni][r] + bv;
                if (rp) v += rp[(size_t)r * N];
                if (relu) v = fmaxf(v, 0.f);
                op[(size_t)r * N] = v;
              }
            }
          }
        } else if (OUTMODE == 1) {
          _Float16* op = outH + (size_t)col * MAPPIX + rbase;
          if (fullM) {
            #pragma unroll
            for (int r = 0; r < 8; ++r) op[r] = (_Float16)acc[mi][ni][r];
          } else {
            #pragma unroll
            for (int r = 0; r < 8; ++r)
              if (rbase + r < M) op[r] = (_Float16)acc[mi][ni][r];
          }
        } else {
          const float bv = bias ? bias[col] : 0.f;
          _Float16* op = outH + (size_t)rbase * N + col;
          #pragma unroll
          for (int r = 0; r < 8; ++r) {
            if (fullM || rbase + r < M) {
              float v = acc[mi][ni][r] + bv;
              if (relu) v = fmaxf(v, 0.f);
              op[(size_t)r * N] = (_Float16)v;
            }
          }
        }
      }
    }
  }
}

// ------------------------------------------------------------------
// Weight converters: f32 -> f16 [N][Kp] (Kp = K rounded up to 32)
// ------------------------------------------------------------------
__global__ void wt_transpose_kernel(const float* __restrict__ W,  // [K][N]
                                    _Float16* __restrict__ Wt, int K, int Kp, int N)
{
  int i = blockIdx.x * blockDim.x + threadIdx.x;
  if (i >= N * Kp) return;
  int n = i / Kp, k = i - n * Kp;
  Wt[i] = (k < K) ? (_Float16)W[(size_t)k * N + n] : (_Float16)0.f;
}

__global__ void wt_cvt_kernel(const float* __restrict__ W,        // [N][K]
                              _Float16* __restrict__ Wt, int K, int Kp, int N)
{
  int i = blockIdx.x * blockDim.x + threadIdx.x;
  if (i >= N * Kp) return;
  int n = i / Kp, k = i - n * Kp;
  Wt[i] = (k < K) ? (_Float16)W[(size_t)n * K + k] : (_Float16)0.f;
}

// ------------------------------------------------------------------
// value = concat(feat.reshape(256,-1).T)  -> [NVAL][256] f16
// ------------------------------------------------------------------
__global__ void build_value_kernel(const float* __restrict__ f0,
                                   const float* __restrict__ f1,
                                   const float* __restrict__ f2,
                                   const float* __restrict__ f3,
                                   _Float16* __restrict__ value)
{
  size_t i = (size_t)blockIdx.x * blockDim.x + threadIdx.x;
  if (i >= (size_t)NVALv * EMBEDv) return;
  int c = (int)(i & 255);
  int r = (int)(i >> 8);
  const float* f; int hw; int s;
  if (r < 32768)       { f = f0; hw = 32768; s = r; }
  else if (r < 40960)  { f = f1; hw = 8192;  s = r - 32768; }
  else if (r < 43008)  { f = f2; hw = 2048;  s = r - 40960; }
  else                 { f = f3; hw = 512;   s = r - 43008; }
  value[i] = (_Float16)f[(size_t)c * hw + s];
}

// ------------------------------------------------------------------
// Sine positional encoding [NQ][256] f32
// ------------------------------------------------------------------
__global__ void pos_kernel(float* __restrict__ pos)
{
  size_t i = (size_t)blockIdx.x * blockDim.x + threadIdx.x;
  if (i >= (size_t)NQv * EMBEDv) return;
  int c = (int)(i & 255);
  int q = (int)(i >> 8);
  int y = q / 150, x = q % 150;
  const float scale = 6.283185307179586f;
  int cc = (c < 128) ? c : c - 128;
  float v = (c < 128) ? (float)(y + 1) : (float)(x + 1);
  v = v / (150.f + 1e-6f) * scale;
  float dim = __powf(10000.f, (float)(cc >> 1) / 64.f);
  float p = v / dim;
  pos[i] = (cc & 1) ? __cosf(p) : __sinf(p);
}

// qpos = (f16)(q + pos)
__global__ void addh_kernel(const float* __restrict__ a,
                            const float* __restrict__ b,
                            _Float16* __restrict__ o, int n)
{
  int i = blockIdx.x * blockDim.x + threadIdx.x;
  if (i < n) o[i] = (_Float16)(a[i] + b[i]);
}

// ------------------------------------------------------------------
// LayerNorm over 256 channels, one wave per row, in-place + f16 mirror
// ------------------------------------------------------------------
__global__ __launch_bounds__(256)
void layernorm_kernel(float* __restrict__ x, const float* __restrict__ g,
                      const float* __restrict__ b, _Float16* __restrict__ xh, int M)
{
  int q = blockIdx.x * 8 + (threadIdx.x >> 5);
  int lane = threadIdx.x & 31;
  if (q >= M) return;
  float* row = x + (size_t)q * 256;
  float vals[8], s = 0.f, s2 = 0.f;
  #pragma unroll
  for (int j = 0; j < 8; ++j) {
    float v = row[lane + 32 * j];
    vals[j] = v; s += v; s2 += v * v;
  }
  #pragma unroll
  for (int o = 16; o; o >>= 1) {
    s  += __shfl_xor(s,  o, 32);
    s2 += __shfl_xor(s2, o, 32);
  }
  float m = s * (1.f / 256.f);
  float var = s2 * (1.f / 256.f) - m * m;
  float r = rsqrtf(var + 1e-5f);
  #pragma unroll
  for (int j = 0; j < 8; ++j) {
    int c = lane + 32 * j;
    float v = (vals[j] - m) * r * g[c] + b[c];
    row[c] = v;
    if (xh) xh[(size_t)q * 256 + c] = (_Float16)v;
  }
}

// ------------------------------------------------------------------
// Deformable sampling: one wave per (query, head); f16 output.
// ------------------------------------------------------------------
__global__ __launch_bounds__(256)
void deform_sample_kernel(const float* __restrict__ vproj,
                          const float* __restrict__ off,
                          const float* __restrict__ att,
                          const float* __restrict__ refp,
                          _Float16* __restrict__ acch)
{
  int q = blockIdx.x;
  int h = threadIdx.x >> 5;
  int lane = threadIdx.x & 31;

  int l = lane >> 3, p = lane & 7, z = p & 3;
  int Wl = 256 >> l, Hl = 128 >> l;
  size_t obase = (size_t)q * 512 + (size_t)((h * 4 + l) * 8 + p) * 2;
  float lx = refp[(size_t)q * 8 + z * 2 + 0] + off[obase + 0] / (float)Wl;
  float ly = refp[(size_t)q * 8 + z * 2 + 1] + off[obase + 1] / (float)Hl;

  float logit = att[(size_t)q * 256 + h * 32 + lane];
  float m = logit;
  #pragma unroll
  for (int o = 16; o; o >>= 1) m = fmaxf(m, __shfl_xor(m, o, 32));
  float e = __expf(logit - m);
  float s = e;
  #pragma unroll
  for (int o = 16; o; o >>= 1) s += __shfl_xor(s, o, 32);
  float w = e / s;

  float a = 0.f;
  for (int pt = 0; pt < 32; ++pt) {
    float wp = __shfl(w,  pt, 32);
    float px = __shfl(lx, pt, 32);
    float py = __shfl(ly, pt, 32);
    int lv = pt >> 3;
    int W = 256 >> lv, H = 128 >> lv;
    int st = (lv == 0) ? 0 : (lv == 1) ? 32768 : (lv == 2) ? 40960 : 43008;
    float xf = px * (float)W - 0.5f;
    float yf = py * (float)H - 0.5f;
    float x0f = floorf(xf), y0f = floorf(yf);
    float wx = xf - x0f, wy = yf - y0f;
    int x0 = (int)x0f, y0 = (int)y0f;
    const float* vb = vproj + (size_t)st * 256 + h * 32 + lane;
    int x0c = x0 < 0 ? 0 : (x0 > W - 1 ? W - 1 : x0);
    int x1c = x0 + 1 < 0 ? 0 : (x0 + 1 > W - 1 ? W - 1 : x0 + 1);
    int y0c = y0 < 0 ? 0 : (y0 > H - 1 ? H - 1 : y0);
    int y1c = y0 + 1 < 0 ? 0 : (y0 + 1 > H - 1 ? H - 1 : y0 + 1);
    bool xv0 = (unsigned)x0 < (unsigned)W;
    bool xv1 = (unsigned)(x0 + 1) < (unsigned)W;
    bool yv0 = (unsigned)y0 < (unsigned)H;
    bool yv1 = (unsigned)(y0 + 1) < (unsigned)H;
    float t00 = vb[(size_t)(y0c * W + x0c) * 256];
    float t01 = vb[(size_t)(y0c * W + x1c) * 256];
    float t10 = vb[(size_t)(y1c * W + x0c) * 256];
    float t11 = vb[(size_t)(y1c * W + x1c) * 256];
    float v00 = (yv0 && xv0) ? t00 : 0.f;
    float v01 = (yv0 && xv1) ? t01 : 0.f;
    float v10 = (yv1 && xv0) ? t10 : 0.f;
    float v11 = (yv1 && xv1) ? t11 : 0.f;
    float sv = v00 * (1.f - wy) * (1.f - wx) + v01 * (1.f - wy) * wx
             + v10 * wy * (1.f - wx)         + v11 * wy * wx;
    a += wp * sv;
  }
  acch[(size_t)q * 256 + h * 32 + lane] = (_Float16)a;
}

// ------------------------------------------------------------------
// Bilinear resize 150x150 -> 500x500 (align_corners), f32 [q][c] -> f16 CHW
// ------------------------------------------------------------------
__global__ void resize_kernel(const float* __restrict__ q, _Float16* __restrict__ bevr)
{
  size_t i = (size_t)blockIdx.x * blockDim.x + threadIdx.x;
  if (i >= (size_t)256 * MAPPIX) return;
  int c   = (int)(i / MAPPIX);
  int pix = (int)(i % MAPPIX);
  int oy = pix / 500, ox = pix % 500;
  const float sc = 149.f / 499.f;
  float fy = oy * sc, fx = ox * sc;
  int y0 = (int)fy; if (y0 > 148) y0 = 148;
  int x0 = (int)fx; if (x0 > 148) x0 = 148;
  float wy = fy - (float)y0, wx = fx - (float)x0;
  const float* base = q + c;
  float v00 = base[(size_t)(y0 * 150 + x0) * 256];
  float v01 = base[(size_t)(y0 * 150 + x0 + 1) * 256];
  float v10 = base[(size_t)((y0 + 1) * 150 + x0) * 256];
  float v11 = base[(size_t)((y0 + 1) * 150 + x0 + 1) * 256];
  float v = (v00 * (1.f - wy) + v10 * wy) * (1.f - wx)
          + (v01 * (1.f - wy) + v11 * wy) * wx;
  bevr[i] = (_Float16)v;
}

// ------------------------------------------------------------------
// Per-channel BN stats over 500x500 (one block per channel, h8 reads)
// ------------------------------------------------------------------
__global__ __launch_bounds__(256)
void stats_kernel(const _Float16* __restrict__ x, float* __restrict__ mean,
                  float* __restrict__ rstd)
{
  int c = blockIdx.x;
  const _Float16* p = x + (size_t)c * MAPPIX;
  float s = 0.f, s2 = 0.f;
  for (int i = threadIdx.x * 8; i < MAPPIX; i += 256 * 8) {
    h8 v = *reinterpret_cast<const h8*>(p + i);
    #pragma unroll
    for (int j = 0; j < 8; ++j) {
      float f = (float)v[j];
      s += f; s2 += f * f;
    }
  }
  __shared__ float sh[2][256];
  sh[0][threadIdx.x] = s; sh[1][threadIdx.x] = s2;
  __syncthreads();
  for (int st = 128; st; st >>= 1) {
    if (threadIdx.x < st) {
      sh[0][threadIdx.x] += sh[0][threadIdx.x + st];
      sh[1][threadIdx.x] += sh[1][threadIdx.x + st];
    }
    __syncthreads();
  }
  if (threadIdx.x == 0) {
    float m = sh[0][0] * (1.f / MAPPIX);
    float var = sh[1][0] * (1.f / MAPPIX) - m * m;
    mean[c] = m;
    rstd[c] = rsqrtf(var + 1e-5f);
  }
}

// ------------------------------------------------------------------
// Final 1x1 conv (48 -> 21) with fused input BN+ReLU, + db5
// ------------------------------------------------------------------
__global__ void conv5_kernel(const _Float16* __restrict__ x,
                             const float* __restrict__ mean,
                             const float* __restrict__ rstd,
                             const float* __restrict__ g,
                             const float* __restrict__ b,
                             const float* __restrict__ w5,
                             const float* __restrict__ b5,
                             float* __restrict__ out)
{
  size_t i = (size_t)blockIdx.x * blockDim.x + threadIdx.x;
  if (i >= (size_t)21 * MAPPIX) return;
  int pix = (int)(i % MAPPIX);
  int n   = (int)(i / MAPPIX);
  float s = b5[n];
  #pragma unroll 4
  for (int c = 0; c < 48; ++c) {
    float v = (float)x[(size_t)c * MAPPIX + pix];
    v = (v - mean[c]) * rstd[c] * g[c] + b[c];
    v = v > 0.f ? v : 0.f;
    s += v * w5[n * 48 + c];
  }
  out[i] = s;
}

// ------------------------------------------------------------------
// observed = proj_indices < max(proj_indices)
// ------------------------------------------------------------------
__global__ void initmax_kernel(int* mx)
{
  if (blockIdx.x == 0 && threadIdx.x == 0) *mx = (-2147483647 - 1);
}

__global__ __launch_bounds__(256)
void maxred_kernel(const int* __restrict__ pi, int n, int* __restrict__ mx)
{
  int i = blockIdx.x * 256 + threadIdx.x;
  int v = (i < n) ? pi[i] : (-2147483647 - 1);
  #pragma unroll
  for (int o = 16; o; o >>= 1) v = max(v, __shfl_xor(v, o, 32));
  if ((threadIdx.x & 31) == 0) atomicMax(mx, v);
}

__global__ void observed_kernel(const int* __restrict__ pi, const int* __restrict__ mx,
                                float* __restrict__ o, int n)
{
  int i = blockIdx.x * blockDim.x + threadIdx.x;
  if (i < n) o[i] = (pi[i] < *mx) ? 1.f : 0.f;
}

// =====================================================================
// Host orchestration
// =====================================================================
extern "C" void kernel_launch(void* const* d_in, const int* in_sizes, int n_in,
                              void* d_out, int out_size, void* d_ws, size_t ws_size,
                              hipStream_t stream)
{
  (void)in_sizes; (void)n_in; (void)out_size; (void)ws_size;

  const float* feat0 = (const float*)d_in[0];
  const float* feat1 = (const float*)d_in[1];
  const float* feat2 = (const float*)d_in[2];
  const float* feat3 = (const float*)d_in[3];
  const float* bevq  = (const float*)d_in[4];
  const float* refp  = (const float*)d_in[5];
  const float* Woff  = (const float*)d_in[6];
  const float* boff  = (const float*)d_in[7];
  const float* Watt  = (const float*)d_in[8];
  const float* batt  = (const float*)d_in[9];
  const float* Wval  = (const float*)d_in[10];
  const float* bval  = (const float*)d_in[11];
  const float* Wout  = (const float*)d_in[12];
  const float* bout  = (const float*)d_in[13];
  const float* g1    = (const float*)d_in[14];
  const float* b1    = (const float*)d_in[15];
  const float* Wf1   = (const float*)d_in[16];
  const float* bf1   = (const float*)d_in[17];
  const float* Wf2   = (const float*)d_in[18];
  const float* bf2   = (const float*)d_in[19];
  const float* g2    = (const float*)d_in[20];
  const float* b2    = (const float*)d_in[21];
  const float* dw1   = (const float*)d_in[22];
  const float* bg1   = (const float*)d_in[23];
  const float* bb1   = (const float*)d_in[24];
  const float* dw2   = (const float*)d_in[25];
  const float* bg2   = (const float*)d_in[26];
  const float* bb2   = (const float*)d_in[27];
  const float* dw3   = (const float*)d_in[28];
  const float* bg3   = (const float*)d_in[29];
  const float* bb3   = (const float*)d_in[30];
  const float* dw4   = (const float*)d_in[31];
  const float* bg4   = (const float*)d_in[32];
  const float* bb4   = (const float*)d_in[33];
  const float* dw5   = (const float*)d_in[34];
  const float* db5   = (const float*)d_in[35];
  const int*   proj  = (const int*)d_in[36];

  float* semmap  = (float*)d_out;                       // [21][250000]
  float* obs_out = (float*)d_out + (size_t)21 * MAPPIX; // [250000]

  // ---- workspace layout (~300 MB) ----
  float* wsf  = (float*)d_ws;
  float* qbuf = wsf;                             // [NQ][256] f32, persistent
  _Float16* wtb = (_Float16*)(wsf + (size_t)NQv * 256);  // 2M halves weight scratch
  const size_t P0 = (size_t)NQv * 256 + 1000000;         // pool base (floats)

  // attention pool
  float* posb = wsf + P0;                          // [NQ][256] f32
  float* q2b  = posb + (size_t)NQv * 256;          // [NQ][256] f32
  float* offb = q2b  + (size_t)NQv * 256;          // [NQ][512] f32
  float* attb = offb + (size_t)NQv * 512;          // [NQ][256] f32
  float* vprj = attb + (size_t)NQv * 256;          // [NVAL][256] f32
  _Float16* qposH = (_Float16*)(vprj + (size_t)NVALv * 256);  // [NQ][256]
  _Float16* accH  = qposH + (size_t)NQv * 256;                // [NQ][256]
  _Float16* q2H   = accH  + (size_t)NQv * 256;                // [NQ][256]
  _Float16* ffnH  = q2H   + (size_t)NQv * 256;                // [NQ][512]
  _Float16* valH  = ffnH  + (size_t)NQv * 512;                // [NVAL][256]

  // conv pool (overlaps attention pool, starts at P0)
  _Float16* bevr = (_Float16*)(wsf + P0);          // [256][500][500]
  _Float16* x1   = bevr + (size_t)256 * MAPPIX;    // [128][...]
  _Float16* x2   = x1   + (size_t)128 * MAPPIX;    // [64][...]
  _Float16* x3   = x2   + (size_t)64  * MAPPIX;    // [48][...]
  _Float16* x4   = x3   + (size_t)48  * MAPPIX;    // [48][...]
  float* st      = (float*)(x4 + (size_t)48 * MAPPIX);
  float* mean1 = st;        float* rstd1 = st + 128;
  float* mean2 = st + 256;  float* rstd2 = st + 320;
  float* mean3 = st + 384;  float* rstd3 = st + 432;
  float* mean4 = st + 480;  float* rstd4 = st + 528;
  int*   mxbuf = (int*)(st + 576);

  // helpers -----------------------------------------------------------
  auto wtT = [&](const float* W, int K, int Kp, int N) {   // [K][N] -> f16 [N][Kp]
    int n = N * Kp;
    wt_transpose_kernel<<<(n + 255) / 256, 256, 0, stream>>>(W, wtb, K, Kp, N);
  };
  auto wtC = [&](const float* W, int K, int Kp, int N) {   // [N][K] -> f16 [N][Kp]
    int n = N * Kp;
    wt_cvt_kernel<<<(n + 255) / 256, 256, 0, stream>>>(W, wtb, K, Kp, N);
  };
  auto gemmF = [&](const _Float16* A, const float* bias, const float* resid,
                   float* out, int M, int N, int K, int relu) {
    dim3 g((M + 127) / 128, (N + 63) / 64);
    wmma_gemm<DenseAH, 0><<<g, 256, 0, stream>>>(
        DenseAH{A, M, K}, wtb, M, N, K, bias, resid, out, nullptr, relu);
  };
  auto gemmH = [&](const _Float16* A, const float* bias, _Float16* out,
                   int M, int N, int K, int relu) {
    dim3 g((M + 127) / 128, (N + 63) / 64);
    wmma_gemm<DenseAH, 2><<<g, 256, 0, stream>>>(
        DenseAH{A, M, K}, wtb, M, N, K, bias, nullptr, nullptr, out, relu);
  };

  // ---- stage 0: value concat + positional encoding + q init ----
  {
    size_t nv = (size_t)NVALv * 256;
    build_value_kernel<<<(unsigned)((nv + 255) / 256), 256, 0, stream>>>(
        feat0, feat1, feat2, feat3, valH);
    size_t np = (size_t)NQv * 256;
    pos_kernel<<<(unsigned)((np + 255) / 256), 256, 0, stream>>>(posb);
    (void)hipMemcpyAsync(qbuf, bevq, np * sizeof(float), hipMemcpyDeviceToDevice, stream);
  }

  // ---- stage 1: two encoder layers ----
  const int nQC = NQv * 256;
  for (int i = 0; i < 2; ++i) {
    const float* WoffL = Woff + (size_t)i * 256 * 512;
    const float* boffL = boff + (size_t)i * 512;
    const float* WattL = Watt + (size_t)i * 256 * 256;
    const float* battL = batt + (size_t)i * 256;
    const float* WvalL = Wval + (size_t)i * 256 * 256;
    const float* bvalL = bval + (size_t)i * 256;
    const float* WoutL = Wout + (size_t)i * 256 * 256;
    const float* boutL = bout + (size_t)i * 256;
    const float* Wf1L  = Wf1  + (size_t)i * 256 * 512;
    const float* bf1L  = bf1  + (size_t)i * 512;
    const float* Wf2L  = Wf2  + (size_t)i * 512 * 256;
    const float* bf2L  = bf2  + (size_t)i * 256;

    addh_kernel<<<(nQC + 255) / 256, 256, 0, stream>>>(qbuf, posb, qposH, nQC);

    wtT(WoffL, 256, 256, 512);
    gemmF(qposH, boffL, nullptr, offb, NQv, 512, 256, 0);
    wtT(WattL, 256, 256, 256);
    gemmF(qposH, battL, nullptr, attb, NQv, 256, 256, 0);
    wtT(WvalL, 256, 256, 256);
    gemmF(valH, bvalL, nullptr, vprj, NVALv, 256, 256, 0);

    deform_sample_kernel<<<NQv, 256, 0, stream>>>(vprj, offb, attb, refp, accH);

    wtT(WoutL, 256, 256, 256);
    gemmF(accH, boutL, qbuf, q2b, NQv, 256, 256, 0);              // + residual q
    layernorm_kernel<<<(NQv + 7) / 8, 256, 0, stream>>>(
        q2b, g1 + i * 256, b1 + i * 256, q2H, NQv);

    wtT(Wf1L, 256, 256, 512);
    gemmH(q2H, bf1L, ffnH, NQv, 512, 256, 1);                     // relu, f16 out
    wtT(Wf2L, 512, 512, 256);
    gemmF(ffnH, bf2L, q2b, qbuf, NQv, 256, 512, 0);               // + residual
    layernorm_kernel<<<(NQv + 7) / 8, 256, 0, stream>>>(
        qbuf, g2 + i * 256, b2 + i * 256, nullptr, NQv);
  }

  // ---- stage 2: resize + conv decoder (implicit-GEMM WMMA) ----
  {
    size_t nr = (size_t)256 * MAPPIX;
    resize_kernel<<<(unsigned)((nr + 255) / 256), 256, 0, stream>>>(qbuf, bevr);
  }
  const int MB = (MAPPIX + 127) / 128;   // 1954 M-tiles

  // conv1: 7x7, 256 -> 128 (K = 12544)
  wtC(dw1, 12544, 12544, 128);
  wmma_gemm<ConvA7H, 1><<<dim3(MB, 2), 256, 0, stream>>>(
      ConvA7H{bevr}, wtb, MAPPIX, 128, 12544, nullptr, nullptr, nullptr, x1, 0);
  stats_kernel<<<128, 256, 0, stream>>>(x1, mean1, rstd1);

  // conv2: 3x3, 128 -> 64 (K = 1152), input BN1+ReLU fused
  wtC(dw2, 1152, 1152, 64);
  wmma_gemm<ConvA3H, 1><<<dim3(MB, 1), 256, 0, stream>>>(
      ConvA3H{x1, mean1, rstd1, bg1, bb1, 128}, wtb,
      MAPPIX, 64, 1152, nullptr, nullptr, nullptr, x2, 0);
  stats_kernel<<<64, 256, 0, stream>>>(x2, mean2, rstd2);

  // conv3: 3x3, 64 -> 48 (K = 576)
  wtC(dw3, 576, 576, 48);
  wmma_gemm<ConvA3H, 1><<<dim3(MB, 1), 256, 0, stream>>>(
      ConvA3H{x2, mean2, rstd2, bg2, bb2, 64}, wtb,
      MAPPIX, 48, 576, nullptr, nullptr, nullptr, x3, 0);
  stats_kernel<<<48, 256, 0, stream>>>(x3, mean3, rstd3);

  // conv4: 3x3, 48 -> 48 (K = 432, padded to 448)
  wtC(dw4, 432, 448, 48);
  wmma_gemm<ConvA3H, 1><<<dim3(MB, 1), 256, 0, stream>>>(
      ConvA3H{x3, mean3, rstd3, bg3, bb3, 48}, wtb,
      MAPPIX, 48, 448, nullptr, nullptr, nullptr, x4, 0);
  stats_kernel<<<48, 256, 0, stream>>>(x4, mean4, rstd4);

  // conv5: 1x1, 48 -> 21 (+db5) -> semmap
  {
    size_t n5 = (size_t)21 * MAPPIX;
    conv5_kernel<<<(unsigned)((n5 + 255) / 256), 256, 0, stream>>>(
        x4, mean4, rstd4, bg4, bb4, dw5, db5, semmap);
  }

  // ---- observed mask ----
  initmax_kernel<<<1, 32, 0, stream>>>(mxbuf);
  maxred_kernel<<<(MAPPIX + 255) / 256, 256, 0, stream>>>(proj, MAPPIX, mxbuf);
  observed_kernel<<<(MAPPIX + 255) / 256, 256, 0, stream>>>(proj, mxbuf, obs_out, MAPPIX);
}
